// DSTP_RNN_37615323579039
// MI455X (gfx1250) — compile-verified, matrix operand order checked by reading, not run on hardware
//
#include <hip/hip_runtime.h>

typedef __attribute__((ext_vector_type(16))) __bf16 v16bf;
typedef __attribute__((ext_vector_type(8)))  float  v8f;

#define WMMA_BF16(a,b,c) __builtin_amdgcn_wmma_f32_16x16x32_bf16(false,(a),false,(b),(short)0,(c),false,false)

namespace {
constexpr int TE=48, NI=18, F1=17, F2=129;
constexpr int BT=16, NWGRID=64, NTHR=512, NWAVE=16;

// ---- LDS byte offsets (dynamic shared, ~297KB < 320KB/WG) ----
constexpr int OFF_BIG = 0;                  // 196608 B : stage-dependent big weights
constexpr int OFF_WE  = 196608;             //  24576 B : context-weight swizzle (K=256,N=48)
constexpr int OFF_HC  = 221184;             //   8192 B : [16][256] bf16  (h | c)
constexpr int OFF_CF  = 229376;             //   8192 B : [16][128] f32 c
constexpr int OFF_HF  = 237568;             //   8192 B : [16][128] f32 h
constexpr int OFF_WO  = 245760;             //   8192 B : we/wd GEMM output (f32, up to 16x128)
constexpr int OFF_SC  = 253952;             //   8448 B : scores [16][132] f32
constexpr int OFF_XS  = 262400;             //   4096 B : [16][128] bf16 gate-GEMM A (x_s / din)
constexpr int OFF_GB  = 266496;             //   2048 B : combined gate bias (512 f32)
constexpr int OFF_WL  = 268544;             //   2048 B : stage2 label column of Wih (512 f32)
constexpr int OFF_VV  = 270592;             //    512 B : V-vector (<=128 f32)
constexpr int OFF_LB  = 271104;             //     64 B : per-b label activations
constexpr int OFF_G   = 271168;             //  32768 B : gates [16][512] f32
constexpr int SMEM_BYTES = OFF_G + 32768;   // 303936
}

__device__ __forceinline__ int kmap(int e, int half) {
  // bf16 16x32 WMMA fragment K-index for element e (0..15) of a lane.
  return ((e >> 3) << 4) + (half << 3) + (((e >> 1) & 3) << 1) + (e & 1);
}

__device__ __forceinline__ v16bf lds_load_a(const __bf16* A, int stride, int k0, int lane) {
  const int m = lane & 15, half = lane >> 4;
  v16bf a;
#pragma unroll
  for (int e = 0; e < 16; ++e) a[e] = A[m * stride + k0 + kmap(e, half)];
  return a;
}

__device__ __forceinline__ v16bf lds_load_bswz(const __bf16* Bs, int frag, int lane) {
  const __bf16* p = Bs + (frag << 9) + (lane << 4);   // 16 contiguous bf16 per lane
  v16bf b;
#pragma unroll
  for (int e = 0; e < 16; ++e) b[e] = p[e];
  return b;
}

__device__ __forceinline__ v16bf glb_load_bfrag(const float* W, int ldK, int Kreal,
                                                int n0, int k0, int lane) {
  const int n = n0 + (lane & 15), half = lane >> 4;
  v16bf b;
#pragma unroll
  for (int e = 0; e < 16; ++e) {
    int k = k0 + kmap(e, half);
    float v = (k < Kreal) ? W[n * ldK + k] : 0.f;
    b[e] = (__bf16)v;
  }
  return b;
}

// Pre-swizzle a [N][K]-major f32 weight into per-fragment bf16 LDS layout.
__device__ inline void swizzleB(__bf16* dst, const float* W, int ldK,
                                int Nreal, int Kreal, int KT, int NT, int tid) {
  const int total = KT * NT * 512;
  for (int idx = tid; idx < total; idx += NTHR) {
    int frag = idx >> 9, r = idx & 511, lane = r >> 4, e = r & 15;
    int kt = frag / NT, nt = frag - kt * NT;
    int n = nt * 16 + (lane & 15);
    int k = kt * 32 + kmap(e, lane >> 4);
    float v = (n < Nreal && k < Kreal) ? W[n * ldK + k] : 0.f;
    dst[idx] = (__bf16)v;
  }
}

// we/wd = [h|c](16x256) @ B(256xN) via WMMA; NT n-tiles spread over 16 waves.
__device__ __forceinline__ void gemm_ctx(const __bf16* hc, const __bf16* Bswz, int NT,
                                         float* Cout, int ldC, int wave, int lane) {
  for (int nt = wave; nt < NT; nt += NWAVE) {
    v8f acc = {};
#pragma unroll
    for (int kf = 0; kf < 8; ++kf) {
      v16bf a = lds_load_a(hc, 256, kf * 32, lane);
      v16bf b = lds_load_bswz(Bswz, kf * NT + nt, lane);
      acc = WMMA_BF16(a, b, acc);
    }
    const int half = lane >> 4, col = nt * 16 + (lane & 15);
#pragma unroll
    for (int r = 0; r < 8; ++r) Cout[(r + 8 * half) * ldC + col] = acc[r];
  }
}

// gates = xs @ Wih.T (LDS-swizzled, KTih k-frags) + h @ Whh.T (VGPR-resident frags)
__device__ __forceinline__ void gemm_gates(const __bf16* xs, const __bf16* hc,
                                           const __bf16* WihSwz, int KTih,
                                           const v16bf (&whh)[2][4],
                                           float* g, int wave, int lane) {
  v16bf ax[4], ah[4];
#pragma unroll
  for (int kf = 0; kf < 4; ++kf) {
    if (kf < KTih) ax[kf] = lds_load_a(xs, 128, kf * 32, lane);
    ah[kf] = lds_load_a(hc, 256, kf * 32, lane);
  }
#pragma unroll
  for (int i = 0; i < 2; ++i) {
    const int nt = wave * 2 + i;
    v8f acc = {};
#pragma unroll
    for (int kf = 0; kf < 4; ++kf)
      if (kf < KTih) acc = WMMA_BF16(ax[kf], lds_load_bswz(WihSwz, kf * 32 + nt, lane), acc);
#pragma unroll
    for (int kf = 0; kf < 4; ++kf) acc = WMMA_BF16(ah[kf], whh[i][kf], acc);
    const int half = lane >> 4, col = nt * 16 + (lane & 15);
#pragma unroll
    for (int r = 0; r < 8; ++r) g[(r + 8 * half) * 512 + col] = acc[r];
  }
}

__device__ __forceinline__ float sigf(float x) { return 1.f / (1.f + expf(-x)); }

__device__ __forceinline__ void cell_update(const float* g, const float* gb,
                                            const float* wlab, const float* labv, int uselab,
                                            float* cF, float* hF, __bf16* hc,
                                            float* gout, int gstride, int tid) {
  for (int it = tid; it < BT * 128; it += NTHR) {
    const int b = it >> 7, j = it & 127;
    const float lv = uselab ? labv[b] : 0.f;
    float gi = g[b * 512 + j]       + gb[j];
    float gf = g[b * 512 + 128 + j] + gb[128 + j];
    float gg = g[b * 512 + 256 + j] + gb[256 + j];
    float go = g[b * 512 + 384 + j] + gb[384 + j];
    if (uselab) { gi += lv * wlab[j]; gf += lv * wlab[128 + j];
                  gg += lv * wlab[256 + j]; go += lv * wlab[384 + j]; }
    const float c2 = sigf(gf) * cF[it] + sigf(gi) * tanhf(gg);
    const float h2 = sigf(go) * tanhf(c2);
    cF[it] = c2; hF[it] = h2;
    hc[b * 256 + j] = (__bf16)h2;
    hc[b * 256 + 128 + j] = (__bf16)c2;
    if (gout) gout[b * gstride + j] = h2;
  }
}

__global__ __launch_bounds__(NTHR, 1)
void dstp_rnn_kernel(
    const float* input_p_q, const float* label_p,
    const float* Ue1_W, const float* Ue1_b, const float* We1_W, const float* Ve1_W, const float* Ve1_b,
    const float* Ue2_W, const float* Ue2_b, const float* We2_W, const float* Ve2_W, const float* Ve2_b,
    const float* Ud_W, const float* Ud_b, const float* Wd_W, const float* Vd_W, const float* Vd_b,
    const float* e1_Wih, const float* e1_Whh, const float* e1_bih, const float* e1_bhh,
    const float* e2_Wih, const float* e2_Whh, const float* e2_bih, const float* e2_bhh,
    const float* d_Wih, const float* d_Whh, const float* d_bih, const float* d_bhh,
    const float* reg_W, const float* reg_b,
    float* out, float* mid, float* pre2, float* fin, float* ud) {
  extern __shared__ char smem[];
  __bf16* sBIG = (__bf16*)(smem + OFF_BIG);
  __bf16* sWih = sBIG;
  __bf16* sPre1 = (__bf16*)(smem + OFF_BIG + 32768);
  __bf16* sWd  = (__bf16*)(smem + OFF_BIG + 131072);
  __bf16* sWe  = (__bf16*)(smem + OFF_WE);
  __bf16* sHC  = (__bf16*)(smem + OFF_HC);
  float*  sCF  = (float*)(smem + OFF_CF);
  float*  sHF  = (float*)(smem + OFF_HF);
  float*  sWO  = (float*)(smem + OFF_WO);
  float*  sSC  = (float*)(smem + OFF_SC);
  __bf16* sXS  = (__bf16*)(smem + OFF_XS);
  float*  sGB  = (float*)(smem + OFF_GB);
  float*  sWL  = (float*)(smem + OFF_WL);
  float*  sVV  = (float*)(smem + OFF_VV);
  float*  sLB  = (float*)(smem + OFF_LB);
  float*  sG   = (float*)(smem + OFF_G);

  const int tid = threadIdx.x, lane = tid & 31, wave = tid >> 5;
  const int bg = blockIdx.x * BT;
  v16bf whh[2][4];

  // ================= Stage 1 =================
  swizzleB(sWih, e1_Wih, 17, 512, 17, 1, 32, tid);
  swizzleB(sWe,  We1_W, 256, 48, 256, 8, 3, tid);
#pragma unroll
  for (int i = 0; i < 2; ++i)
#pragma unroll
    for (int kf = 0; kf < 4; ++kf)
      whh[i][kf] = glb_load_bfrag(e1_Whh, 128, 128, (wave * 2 + i) * 16, kf * 32, lane);
  for (int j = tid; j < 512; j += NTHR) sGB[j] = e1_bih[j] + e1_bhh[j];
  for (int j = tid; j < 48; j += NTHR) sVV[j] = Ve1_W[j];
  for (int j = tid; j < BT * 128; j += NTHR) { sCF[j] = 0.f; sHF[j] = 0.f; }
  for (int j = tid; j < BT * 256; j += NTHR) sHC[j] = (__bf16)0.f;
  for (int j = tid; j < BT * 128; j += NTHR) sXS[j] = (__bf16)0.f;   // zero pad cols >=17
  for (int it = tid; it < BT * F1 * TE; it += NTHR) {                // pre1 (bf16 in LDS)
    const int b = it / (F1 * TE), r = it - b * (F1 * TE), f = r / TE, s = r - f * TE;
    float acc = Ue1_b[s];
    for (int t = 0; t < TE; ++t)
      acc += input_p_q[(bg + b) * TE * NI + t * NI + 1 + f] * Ue1_W[s * TE + t];
    sPre1[(b * F1 + f) * TE + s] = (__bf16)acc;
  }
  __syncthreads();

  for (int t = 0; t < TE; ++t) {
    gemm_ctx(sHC, sWe, 3, sWO, 48, wave, lane);
    __syncthreads();
    for (int it = tid; it < BT * F1; it += NTHR) {
      const int b = it / F1, f = it - b * F1;
      float acc = Ve1_b[0];
      for (int s = 0; s < TE; ++s)
        acc += tanhf(sWO[b * 48 + s] + (float)sPre1[(b * F1 + f) * TE + s]) * sVV[s];
      sSC[b * 132 + f] = acc;
    }
    __syncthreads();
    if (tid < BT) {
      const int b = tid;
      float mx = -1e30f;
      for (int f = 0; f < F1; ++f) mx = fmaxf(mx, sSC[b * 132 + f]);
      float sum = 0.f;
      for (int f = 0; f < F1; ++f) { float e = expf(sSC[b * 132 + f] - mx); sSC[b * 132 + f] = e; sum += e; }
      const float inv = 1.f / sum;
      for (int f = 0; f < F1; ++f)
        sXS[b * 128 + f] = (__bf16)(input_p_q[(bg + b) * TE * NI + t * NI + 1 + f] * sSC[b * 132 + f] * inv);
    }
    __syncthreads();
    gemm_gates(sXS, sHC, sWih, 1, whh, sG, wave, lane);
    __syncthreads();
    cell_update(sG, sGB, nullptr, nullptr, 0, sCF, sHF, sHC,
                mid + bg * TE * F2 + t * F2, TE * F2, tid);
    __syncthreads();
  }

  // ================= Stage 2 =================
  __threadfence(); __syncthreads();
  for (int it = tid; it < BT * TE; it += NTHR) {                      // label feature
    const int b = it / TE, t = it - b * TE;
    mid[(bg + b) * TE * F2 + t * F2 + 128] = label_p[(bg + b) * TE + t];
  }
  __threadfence(); __syncthreads();
  for (int it = tid; it < BT * F2 * TE; it += NTHR) {                 // pre2 -> ws (L2)
    const int b = it / (F2 * TE), r = it - b * (F2 * TE), f = r / TE, s = r - f * TE;
    float acc = Ue2_b[s];
    const float* mrow = mid + (bg + b) * TE * F2;
    for (int t = 0; t < TE; ++t) acc += mrow[t * F2 + f] * Ue2_W[s * TE + t];
    pre2[(bg + b) * F2 * TE + f * TE + s] = acc;
  }
  swizzleB(sWih, e2_Wih, 129, 512, 128, 4, 32, tid);
  swizzleB(sWe,  We2_W, 256, 48, 256, 8, 3, tid);
#pragma unroll
  for (int i = 0; i < 2; ++i)
#pragma unroll
    for (int kf = 0; kf < 4; ++kf)
      whh[i][kf] = glb_load_bfrag(e2_Whh, 128, 128, (wave * 2 + i) * 16, kf * 32, lane);
  for (int j = tid; j < 512; j += NTHR) { sGB[j] = e2_bih[j] + e2_bhh[j]; sWL[j] = e2_Wih[j * 129 + 128]; }
  for (int j = tid; j < 48; j += NTHR) sVV[j] = Ve2_W[j];
  for (int j = tid; j < BT * 128; j += NTHR) { sCF[j] = 0.f; sHF[j] = 0.f; }
  for (int j = tid; j < BT * 256; j += NTHR) sHC[j] = (__bf16)0.f;
  __threadfence(); __syncthreads();

  for (int t = 0; t < TE; ++t) {
    gemm_ctx(sHC, sWe, 3, sWO, 48, wave, lane);
    __syncthreads();
    for (int it = tid; it < BT * F2; it += NTHR) {
      const int b = it / F2, f = it - b * F2;
      float acc = Ve2_b[0];
      const float* p2 = pre2 + (bg + b) * F2 * TE + f * TE;
      for (int s = 0; s < TE; ++s) acc += tanhf(sWO[b * 48 + s] + p2[s]) * sVV[s];
      sSC[b * 132 + f] = acc;
    }
    __syncthreads();
    if (tid < BT) {
      const int b = tid;
      float mx = -1e30f;
      for (int f = 0; f < F2; ++f) mx = fmaxf(mx, sSC[b * 132 + f]);
      float sum = 0.f;
      for (int f = 0; f < F2; ++f) { float e = expf(sSC[b * 132 + f] - mx); sSC[b * 132 + f] = e; sum += e; }
      const float inv = 1.f / sum;
      const float* mrow = mid + (bg + b) * TE * F2 + t * F2;
      for (int f = 0; f < F2; ++f) {
        const float xv = mrow[f] * sSC[b * 132 + f] * inv;
        if (f < 128) sXS[b * 128 + f] = (__bf16)xv; else sLB[b] = xv;
      }
    }
    __syncthreads();
    gemm_gates(sXS, sHC, sWih, 4, whh, sG, wave, lane);
    __syncthreads();
    cell_update(sG, sGB, sWL, sLB, 1, sCF, sHF, sHC,
                fin + bg * TE * 128 + t * 128, TE * 128, tid);
    __syncthreads();
  }

  // ================= Decoder =================
  __threadfence(); __syncthreads();
  for (int it = tid; it < BT * TE * 128; it += NTHR) {                // ud -> ws (L2)
    const int b = it / (TE * 128), r = it - b * (TE * 128), t = r >> 7, j = r & 127;
    float acc = Ud_b[j];
    const float* frow = fin + (bg + b) * TE * 128 + t * 128;
    for (int k = 0; k < 128; ++k) acc += frow[k] * Ud_W[j * 128 + k];
    ud[(bg + b) * TE * 128 + t * 128 + j] = acc;
  }
  swizzleB(sWih, d_Wih, 128, 512, 128, 4, 32, tid);
  swizzleB(sWd,  Wd_W, 256, 128, 256, 8, 8, tid);
#pragma unroll
  for (int i = 0; i < 2; ++i)
#pragma unroll
    for (int kf = 0; kf < 4; ++kf)
      whh[i][kf] = glb_load_bfrag(d_Whh, 128, 128, (wave * 2 + i) * 16, kf * 32, lane);
  for (int j = tid; j < 512; j += NTHR) sGB[j] = d_bih[j] + d_bhh[j];
  for (int j = tid; j < 128; j += NTHR) sVV[j] = Vd_W[j];
  for (int j = tid; j < BT * 128; j += NTHR) { sCF[j] = 0.f; sHF[j] = 0.f; }
  for (int j = tid; j < BT * 256; j += NTHR) sHC[j] = (__bf16)0.f;
  __threadfence(); __syncthreads();

  for (int s = 0; s < 30; ++s) {
    gemm_ctx(sHC, sWd, 8, sWO, 128, wave, lane);
    __syncthreads();
    for (int it = tid; it < BT * TE; it += NTHR) {
      const int b = it / TE, t = it - b * TE;
      float acc = Vd_b[0];
      const float* urow = ud + (bg + b) * TE * 128 + t * 128;
      for (int h = 0; h < 128; ++h) acc += tanhf(sWO[b * 128 + h] + urow[h]) * sVV[h];
      sSC[b * 132 + t] = acc;
    }
    __syncthreads();
    if (tid < BT) {
      const int b = tid;
      float mx = -1e30f;
      for (int t = 0; t < TE; ++t) mx = fmaxf(mx, sSC[b * 132 + t]);
      float sum = 0.f;
      for (int t = 0; t < TE; ++t) { float e = expf(sSC[b * 132 + t] - mx); sSC[b * 132 + t] = e; sum += e; }
      const float inv = 1.f / sum;
      for (int t = 0; t < TE; ++t) sSC[b * 132 + t] *= inv;
    }
    __syncthreads();
    for (int it = tid; it < BT * 128; it += NTHR) {                   // din = soft @ fin
      const int b = it >> 7, j = it & 127;
      float acc = 0.f;
      const float* fb = fin + (bg + b) * TE * 128;
      for (int t = 0; t < TE; ++t) acc += sSC[b * 132 + t] * fb[t * 128 + j];
      sXS[b * 128 + j] = (__bf16)acc;
    }
    __syncthreads();
    gemm_gates(sXS, sHC, sWih, 4, whh, sG, wave, lane);
    __syncthreads();
    cell_update(sG, sGB, nullptr, nullptr, 0, sCF, sHF, sHC, nullptr, 0, tid);
    __syncthreads();
    if (s >= 6 && tid < BT) {
      const int b = tid;
      float acc = reg_b[0];
      for (int j = 0; j < 128; ++j) acc += sHF[b * 128 + j] * reg_W[j];
      out[(bg + b) * 24 + (s - 6)] = acc;
    }
    __syncthreads();
  }
}

extern "C" void kernel_launch(void* const* d_in, const int* in_sizes, int n_in,
                              void* d_out, int out_size, void* d_ws, size_t ws_size,
                              hipStream_t stream) {
  const float* in[31];
  for (int i = 0; i < 31; ++i) in[i] = (const float*)d_in[i];
  float* wsf  = (float*)d_ws;
  float* mid  = wsf;                 // 1024*48*129
  float* pre2 = wsf + 6340608;       // 1024*129*48
  float* fin  = wsf + 12681216;      // 1024*48*128
  float* ud   = wsf + 18972672;      // 1024*48*128
  (void)in_sizes; (void)n_in; (void)out_size; (void)ws_size;

  hipFuncSetAttribute(reinterpret_cast<const void*>(dstp_rnn_kernel),
                      hipFuncAttributeMaxDynamicSharedMemorySize, SMEM_BYTES);

  dstp_rnn_kernel<<<NWGRID, NTHR, SMEM_BYTES, stream>>>(
      in[0], in[1], in[2], in[3], in[4], in[5], in[6], in[7], in[8], in[9],
      in[10], in[11], in[12], in[13], in[14], in[15], in[16], in[17], in[18], in[19],
      in[20], in[21], in[22], in[23], in[24], in[25], in[26], in[27], in[28], in[29], in[30],
      (float*)d_out, mid, pre2, fin, ud);
}
